// srm_snn_47029891891491
// MI455X (gfx1250) — matrix operand order repaired and checked by reading, not compiled, not run on hardware
//
#include <hip/hip_runtime.h>
#include <hip/hip_bf16.h>

// ---------------------------------------------------------------------------
// SRM spiking CNN forward for gfx1250 (MI455X), wave32.
// conv/fc as one unified GEMM: D = Wpad[Mpad x Kpad] * Bm[N x Kpad]^T
//   - weights zero-padded at pack time  -> branch-free A fragment loads
//   - im2col materialized patch-major   -> branch-free B fragment loads
//   - v_wmma_f32_16x16x32_f16, 2 N-tiles per wave (A reuse)
//   - SRM neuron update fused in epilogue
// ---------------------------------------------------------------------------

typedef __attribute__((ext_vector_type(16))) _Float16 v16h;
typedef __attribute__((ext_vector_type(8)))  _Float16 v8h;
typedef __attribute__((ext_vector_type(8)))  float    v8f;

#define SRM_THRESH  0.3f
#define SRM_DECAY_M 0.93941306281347578f   // exp(-1/16)
#define SRM_DECAY_S 0.77880078307140487f   // exp(-1/4)

#define BATCH 16
#define TSTEPS 16

// ---------------------------------------------------------------------------
// Pack weights f32 [Mreal,K] -> f16 [Mpad,Kpad], zero-filled padding.
// ---------------------------------------------------------------------------
__global__ void pack_weights_kernel(const float* __restrict__ in,
                                    _Float16* __restrict__ out,
                                    int Mreal, int K, int Mpad, int Kpad) {
  long total = (long)Mpad * Kpad;
  long i = blockIdx.x * (long)blockDim.x + threadIdx.x;
  if (i >= total) return;
  int k = (int)(i % Kpad);
  int m = (int)(i / Kpad);
  _Float16 v = (_Float16)0.0f;
  if (m < Mreal && k < K) v = (_Float16)in[(long)m * K + k];
  out[i] = v;
}

// ---------------------------------------------------------------------------
// Input slice: clip(input[:, t], 0, 1) -> f16  [B,2,64,64]
// ---------------------------------------------------------------------------
__global__ void clip_input_kernel(const float* __restrict__ in,
                                  _Float16* __restrict__ out, int t) {
  const int per = 2 * 64 * 64;
  int i = blockIdx.x * blockDim.x + threadIdx.x;
  if (i >= BATCH * per) return;
  int b = i / per, r = i - b * per;
  float v = in[((long)b * TSTEPS + t) * per + r];
  out[i] = (_Float16)fminf(fmaxf(v, 0.0f), 1.0f);
}

// ---------------------------------------------------------------------------
// im2col (3x3, pad 1): spikes [B,Cin,H,W] -> patch-major Bm [B*H*W][Kpad].
// Owns all boundary predication so the GEMM loop is branch-free.
// ---------------------------------------------------------------------------
__global__ void im2col_kernel(const _Float16* __restrict__ in,
                              _Float16* __restrict__ Bm,
                              int Cin, int H, int W, int K, int Kpad) {
  long total = (long)BATCH * H * W * Kpad;
  long i = blockIdx.x * (long)blockDim.x + threadIdx.x;
  if (i >= total) return;
  int k = (int)(i % Kpad);
  long n = i / Kpad;
  _Float16 val = (_Float16)0.0f;
  if (k < K) {
    int HW = H * W;
    int b = (int)(n / HW);
    int rem = (int)(n - (long)b * HW);
    int y = rem / W, x = rem - y * W;
    int ic = k / 9, r9 = k - ic * 9;
    int ky = r9 / 3, kx = r9 - ky * 3;
    int yy = y + ky - 1, xx = x + kx - 1;
    if ((unsigned)yy < (unsigned)H && (unsigned)xx < (unsigned)W)
      val = in[(((long)b * Cin + ic) * H + yy) * W + xx];
  }
  Bm[i] = val;
}

// ---------------------------------------------------------------------------
// 2x2 max-pool on f16 spikes, NCHW
// ---------------------------------------------------------------------------
__global__ void maxpool2_kernel(const _Float16* __restrict__ in,
                                _Float16* __restrict__ out,
                                int C, int Hout, int Wout) {
  int total = BATCH * C * Hout * Wout;
  int i = blockIdx.x * blockDim.x + threadIdx.x;
  if (i >= total) return;
  int xo = i % Wout; int t = i / Wout;
  int yo = t % Hout; t /= Hout;                 // t = b*C + c
  int Win = Wout * 2;
  const _Float16* p = in + (((long)t * Hout * 2) + yo * 2) * Win + xo * 2;
  float a = (float)p[0], b = (float)p[1];
  float c = (float)p[Win], d = (float)p[Win + 1];
  out[i] = (_Float16)fmaxf(fmaxf(a, b), fmaxf(c, d));
}

// ---------------------------------------------------------------------------
// Fragment loaders: contiguous vector loads, same k order for A and B.
//   A 16x32 f16: lane(hi,lo) halves 0..15 -> K = kbase + hi*16 + 0..15
//   B 32x16 f16: lane(hi,lo) halves 0..15 -> K = kbase + hi*16 + 0..15
// ---------------------------------------------------------------------------
__device__ __forceinline__ v16h load_afrag(const _Float16* __restrict__ wrow,
                                           int kbase, int hi) {
  v8h a0 = *(const v8h*)(wrow + kbase + hi * 16);
  v8h a1 = *(const v8h*)(wrow + kbase + hi * 16 + 8);
  v16h r;
#pragma unroll
  for (int i = 0; i < 8; ++i) { r[i] = a0[i]; r[8 + i] = a1[i]; }
  return r;
}

__device__ __forceinline__ v16h load_bfrag(const _Float16* __restrict__ brow,
                                           int kbase, int hi) {
  return *(const v16h*)(brow + kbase + hi * 16);
}

// ---------------------------------------------------------------------------
// SRM neuron update
// ---------------------------------------------------------------------------
__device__ __forceinline__ float srm_step(float I, float* vm, float* vs,
                                          _Float16* sp, long idx) {
  float sprev = (float)sp[idx];
  float keep  = 1.0f - sprev;
  float nvm = SRM_DECAY_M * vm[idx] * keep + I;
  float nvs = SRM_DECAY_S * vs[idx] * keep + I;
  vm[idx] = nvm;
  vs[idx] = nvs;
  float s = ((nvm - nvs) > SRM_THRESH) ? 1.0f : 0.0f;
  sp[idx] = (_Float16)s;
  return s;
}

// ---------------------------------------------------------------------------
// Unified GEMM + fused SRM.
//   Bm: [N][Kpad] f16 (N = BATCH * HW; HW = spatial size, 1 for FC)
//   Wh: [Mpad][Kpad] f16, zero padded
//   State layout: conv -> [B][Cstride][HW] with Cstride = Cout
//                 fc   -> [B][Cstride]     with HW = 1, Cstride = ldS
//   hwShift = log2(HW).  accum: out_sum[b*11 + j] += spike (fc4 only).
// One wave per (m_tile, pair of n_tiles); A fragment reused across N-tiles.
// ---------------------------------------------------------------------------
__global__ void gemm_srm_wmma_kernel(const _Float16* __restrict__ Bm,
                                     const _Float16* __restrict__ Wh,
                                     float* __restrict__ vm,
                                     float* __restrict__ vs,
                                     _Float16* __restrict__ sp,
                                     int Kpad, int Mreal, int Mpad,
                                     int HW, int hwShift, int Cstride,
                                     float* __restrict__ accum) {
  const int tiles_m = Mpad >> 4;
  const int tiles_n = (BATCH * HW) >> 4;
  const int npairs  = (tiles_n + 1) >> 1;
  int wid = blockIdx.x * (blockDim.x >> 5) + (threadIdx.x >> 5);
  if (wid >= tiles_m * npairs) return;          // wave-uniform
  const int m_tile = wid % tiles_m;
  const int nt0 = (wid / tiles_m) * 2;
  const bool has2 = (nt0 + 1) < tiles_n;        // wave-uniform

  const int lane = threadIdx.x & 31;
  const int hi = lane >> 4;
  const int lo = lane & 15;

  const _Float16* wrow  = Wh + (long)(m_tile * 16 + lo) * Kpad;
  const _Float16* brow0 = Bm + (long)(nt0 * 16 + lo) * Kpad;
  const _Float16* brow1 = brow0 + (long)16 * Kpad;

  v8f acc0 = {0.f, 0.f, 0.f, 0.f, 0.f, 0.f, 0.f, 0.f};
  v8f acc1 = {0.f, 0.f, 0.f, 0.f, 0.f, 0.f, 0.f, 0.f};
  const int ksteps = Kpad >> 5;

  if (has2) {
    for (int ks = 0; ks < ksteps; ++ks) {
      int kbase = ks << 5;
      v16h a  = load_afrag(wrow, kbase, hi);
      v16h b0 = load_bfrag(brow0, kbase, hi);
      v16h b1 = load_bfrag(brow1, kbase, hi);
      acc0 = __builtin_amdgcn_wmma_f32_16x16x32_f16(false, a, false, b0,
                                                    (short)0, acc0, false, false);
      acc1 = __builtin_amdgcn_wmma_f32_16x16x32_f16(false, a, false, b1,
                                                    (short)0, acc1, false, false);
    }
  } else {
    for (int ks = 0; ks < ksteps; ++ks) {
      int kbase = ks << 5;
      v16h a  = load_afrag(wrow, kbase, hi);
      v16h b0 = load_bfrag(brow0, kbase, hi);
      acc0 = __builtin_amdgcn_wmma_f32_16x16x32_f16(false, a, false, b0,
                                                    (short)0, acc0, false, false);
    }
  }

  // ---- fused SRM epilogue: lane owns rows M = r + 8*hi, column N = lo ----
  const int hwMask = HW - 1;
#pragma unroll
  for (int r = 0; r < 8; ++r) {
    int oc = m_tile * 16 + r + 8 * hi;
    if (oc < Mreal) {
      int n = nt0 * 16 + lo;
      int b = n >> hwShift, rem = n & hwMask;
      long idx = ((long)b * Cstride + oc) * HW + rem;
      float s = srm_step(acc0[r], vm, vs, sp, idx);
      if (accum != nullptr && oc < 11) accum[b * 11 + oc] += s;
    }
  }
  if (has2) {
#pragma unroll
    for (int r = 0; r < 8; ++r) {
      int oc = m_tile * 16 + r + 8 * hi;
      if (oc < Mreal) {
        int n = (nt0 + 1) * 16 + lo;
        int b = n >> hwShift, rem = n & hwMask;
        long idx = ((long)b * Cstride + oc) * HW + rem;
        float s = srm_step(acc1[r], vm, vs, sp, idx);
        if (accum != nullptr && oc < 11) accum[b * 11 + oc] += s;
      }
    }
  }
}

// ---------------------------------------------------------------------------
// Final: out = accum / T
// ---------------------------------------------------------------------------
__global__ void finalize_kernel(const float* __restrict__ accum,
                                float* __restrict__ out, int n, float scale) {
  int i = blockIdx.x * blockDim.x + threadIdx.x;
  if (i < n) out[i] = accum[i] * scale;
}

// ---------------------------------------------------------------------------
// Host launch
// ---------------------------------------------------------------------------
extern "C" void kernel_launch(void* const* d_in, const int* in_sizes, int n_in,
                              void* d_out, int out_size, void* d_ws, size_t ws_size,
                              hipStream_t stream) {
  const float* inp = (const float*)d_in[0];   // [16,16,2,64,64]
  const float* W0  = (const float*)d_in[1];   // [64,2,3,3]
  const float* W1  = (const float*)d_in[2];   // [128,64,3,3]
  const float* W2  = (const float*)d_in[3];   // [128,128,3,3]
  const float* W3  = (const float*)d_in[4];   // [512,32768]
  const float* W4  = (const float*)d_in[5];   // [11,512]
  float* out = (float*)d_out;                 // [16,11]

  // -------- state sizes (elements) --------
  const size_t N0 = (size_t)BATCH * 64 * 64 * 64;
  const size_t N1 = (size_t)BATCH * 128 * 64 * 64;
  const size_t N2 = (size_t)BATCH * 128 * 32 * 32;
  const size_t N3 = (size_t)BATCH * 512;
  const size_t N4 = (size_t)BATCH * 16;
  const size_t S  = N0 + N1 + N2 + N3 + N4;
  const size_t OFF0 = 0, OFF1 = N0, OFF2 = N0 + N1, OFF3 = N0 + N1 + N2,
               OFF4 = N0 + N1 + N2 + N3;

  // packed weight sizes [Mpad][Kpad]
  const size_t nW0 = (size_t)64 * 32;        // K=18  -> 32
  const size_t nW1 = (size_t)128 * 576;
  const size_t nW2 = (size_t)128 * 1152;
  const size_t nW3 = (size_t)512 * 32768;
  const size_t nW4 = (size_t)16 * 512;       // M=11 -> 16
  const size_t nBm = (size_t)BATCH * 64 * 64 * 576;   // largest im2col (conv1)

  size_t o = 0;
  auto take = [&](size_t bytes) { size_t r = o; o = (o + bytes + 255) & ~(size_t)255; return r; };
  char* ws = (char*)d_ws;
  float*    vm      = (float*)   (ws + take(S * 4));
  float*    vs      = (float*)   (ws + take(S * 4));
  _Float16* sp      = (_Float16*)(ws + take(S * 2));
  _Float16* in_sp   = (_Float16*)(ws + take((size_t)BATCH * 2 * 64 * 64 * 2));
  _Float16* p1      = (_Float16*)(ws + take(N2 * 2));
  _Float16* p2      = (_Float16*)(ws + take((size_t)BATCH * 32768 * 2));
  _Float16* Bm      = (_Float16*)(ws + take(nBm * 2));
  _Float16* W0h     = (_Float16*)(ws + take(nW0 * 2));
  _Float16* W1h     = (_Float16*)(ws + take(nW1 * 2));
  _Float16* W2h     = (_Float16*)(ws + take(nW2 * 2));
  _Float16* W3h     = (_Float16*)(ws + take(nW3 * 2));
  _Float16* W4h     = (_Float16*)(ws + take(nW4 * 2));
  float*    out_sum = (float*)   (ws + take((size_t)BATCH * 11 * 4));
  (void)ws_size;

  // -------- zero persistent state every launch (deterministic) --------
  hipMemsetAsync(vm, 0, S * 4, stream);
  hipMemsetAsync(vs, 0, S * 2 * 2, stream);
  hipMemsetAsync(sp, 0, S * 2, stream);
  hipMemsetAsync(out_sum, 0, (size_t)BATCH * 11 * 4, stream);

  // -------- pack weights (zero-padded) --------
  auto pack = [&](const float* src, _Float16* dst, int Mreal, int K,
                  int Mpad, int Kpad) {
    long n = (long)Mpad * Kpad;
    pack_weights_kernel<<<(int)((n + 255) / 256), 256, 0, stream>>>(
        src, dst, Mreal, K, Mpad, Kpad);
  };
  pack(W0, W0h, 64, 18, 64, 32);
  pack(W1, W1h, 128, 576, 128, 576);
  pack(W2, W2h, 128, 1152, 128, 1152);
  pack(W3, W3h, 512, 32768, 512, 32768);
  pack(W4, W4h, 11, 512, 16, 512);

  auto gemm = [&](const _Float16* B, const _Float16* W, size_t off,
                  int Kpad, int Mreal, int Mpad, int HW, int hwShift,
                  int Cstride, float* accum) {
    int tiles_m = Mpad >> 4;
    int tiles_n = (BATCH * HW) >> 4;
    int units = tiles_m * ((tiles_n + 1) >> 1);
    gemm_srm_wmma_kernel<<<(units + 7) / 8, 256, 0, stream>>>(
        B, W, vm + off, vs + off, sp + off, Kpad, Mreal, Mpad, HW, hwShift,
        Cstride, accum);
  };
  auto im2col = [&](const _Float16* src, int Cin, int H, int W, int K, int Kpad) {
    long n = (long)BATCH * H * W * Kpad;
    im2col_kernel<<<(int)((n + 255) / 256), 256, 0, stream>>>(
        src, Bm, Cin, H, W, K, Kpad);
  };

  // -------- time loop (sequential state carry) --------
  for (int t = 0; t < TSTEPS; ++t) {
    {
      int n = BATCH * 2 * 64 * 64;
      clip_input_kernel<<<(n + 255) / 256, 256, 0, stream>>>(inp, in_sp, t);
    }
    // L0: conv 2->64 @ 64x64
    im2col(in_sp, 2, 64, 64, 18, 32);
    gemm(Bm, W0h, OFF0, 32, 64, 64, 4096, 12, 64, nullptr);
    // L1: conv 64->128 @ 64x64
    im2col(sp + OFF0, 64, 64, 64, 576, 576);
    gemm(Bm, W1h, OFF1, 576, 128, 128, 4096, 12, 128, nullptr);
    // pool s1 -> p1 [16,128,32,32]
    {
      int n = BATCH * 128 * 32 * 32;
      maxpool2_kernel<<<(n + 255) / 256, 256, 0, stream>>>(sp + OFF1, p1, 128, 32, 32);
    }
    // L2: conv 128->128 @ 32x32
    im2col(p1, 128, 32, 32, 1152, 1152);
    gemm(Bm, W2h, OFF2, 1152, 128, 128, 1024, 10, 128, nullptr);
    // pool s2 -> p2 == flatten [16,32768]
    {
      int n = BATCH * 128 * 16 * 16;
      maxpool2_kernel<<<(n + 255) / 256, 256, 0, stream>>>(sp + OFF2, p2, 128, 16, 16);
    }
    // L3: fc 32768 -> 512   (B = p2 directly, [16][32768])
    gemm(p2, W3h, OFF3, 32768, 512, 512, 1, 0, 512, nullptr);
    // L4: fc 512 -> 11 (padded to 16), accumulate spikes
    gemm(sp + OFF3, W4h, OFF4, 512, 11, 16, 1, 0, 16, out_sum);
  }

  finalize_kernel<<<1, 256, 0, stream>>>(out_sum, out, BATCH * 11, 1.0f / TSTEPS);
}
